// CustomMHA_19207093748249
// MI455X (gfx1250) — matrix operand order repaired and checked by reading, not compiled, not run on hardware
//
#include <hip/hip_runtime.h>
#include <hip/hip_bf16.h>
#include <math.h>

typedef __attribute__((ext_vector_type(16))) _Float16 v16h;
typedef __attribute__((ext_vector_type(8)))  _Float16 v8h;
typedef __attribute__((ext_vector_type(4)))  _Float16 v4h;
typedef __attribute__((ext_vector_type(8)))  float    v8f;

#define B_  2
#define S_  2048
#define E_  1024
#define H_  16
#define HD_ 64
#define N3E (3 * E_)
#define M_  (B_ * S_)

__device__ __forceinline__ v8f wmma_f16(const v16h a, const v16h b, const v8f c) {
  // D = A(16x32 f16) x B(32x16 f16) + C(16x16 f32)
  return __builtin_amdgcn_wmma_f32_16x16x32_f16(false, a, false, b, (short)0, c,
                                                false, false);
}

__device__ __forceinline__ v16h cat16(v8h lo, v8h hi) {
  return __builtin_shufflevector(lo, hi, 0, 1, 2, 3, 4, 5, 6, 7, 8, 9, 10, 11,
                                 12, 13, 14, 15);
}

// CDNA5 async global->LDS copy (tracked on ASYNCcnt, no VGPR round-trip).
// LDS address comes from the dst VGPR; AS(3) offsets are zero-based, and the low
// 32 bits of a generic LDS pointer are exactly that offset.
__device__ __forceinline__ void async_ld_b128(const void* lds_dst,
                                              const void* gsrc) {
  const unsigned int loff = (unsigned int)(uintptr_t)lds_dst;
  const unsigned long long ga = (unsigned long long)(uintptr_t)gsrc;
  asm volatile("global_load_async_to_lds_b128 %0, %1, off"
               :
               : "v"(loff), "v"(ga)
               : "memory");
}
__device__ __forceinline__ void async_wait0() {
  asm volatile("s_wait_asynccnt 0x0" ::: "memory");
}

// -------------------------------------------------------------------------
// Kernel 0: elementwise f32 -> f16 convert (query and qkv_w), 4 elems/thread
// -------------------------------------------------------------------------
__global__ __launch_bounds__(256) void cvt_f32_f16(const float* __restrict__ s,
                                                   _Float16* __restrict__ d,
                                                   int n4) {
  const int i = blockIdx.x * 256 + threadIdx.x;
  if (i >= n4) return;
  const float4 v = ((const float4*)s)[i];
  v4h o;
  o[0] = (_Float16)v.x;
  o[1] = (_Float16)v.y;
  o[2] = (_Float16)v.z;
  o[3] = (_Float16)v.w;
  ((v4h*)d)[i] = o;
}

// -------------------------------------------------------------------------
// Kernel 1: fused QKV projection qkv = query @ W^T + b (M=4096,N=3072,K=1024)
// Block = 8 waves sharing one 64-col W block async-staged into LDS (double
// buffered). Each wave owns a 32(M) x 64(N) tile: 2 A fragments reuse every
// B fragment -> 8 WMMAs per 32-deep k-step. Outputs scattered to
// Qh[b,h,s,d], Kh[b,h,s,d], Vt[b,h,d,s] (V transposed for attention).
// -------------------------------------------------------------------------
__global__ __launch_bounds__(256) void qkv_gemm_kernel(
    const _Float16* __restrict__ Qf, const _Float16* __restrict__ Wh,
    const float* __restrict__ bias, _Float16* __restrict__ Qh,
    _Float16* __restrict__ Kh, _Float16* __restrict__ Vt) {
  __shared__ _Float16 lB[2][64 * 32];            // 2 x 4KB W chunk
  const int tid  = threadIdx.x;
  const int lane = tid & 31;
  const int wave = tid >> 5;
  const int NT   = N3E / 64;                     // 48 column blocks
  const int n0   = (blockIdx.x % NT) * 64;
  const int m0   = ((blockIdx.x / NT) * 8 + wave) * 32;
  const int lr   = lane & 15;
  const int hs   = lane >> 4;

  v8f acc[2][4] = {{{}, {}, {}, {}}, {{}, {}, {}, {}}};
  const _Float16* arow0 = Qf + (size_t)(m0 + lr) * E_;
  const _Float16* arow1 = arow0 + (size_t)16 * E_;

  // cooperative async stage: rows n0..n0+63, 32 halves each at offset kb
  auto stage = [&](int buf, int kb) {
    const _Float16* src = Wh + (size_t)(n0 + (tid >> 2)) * E_ + kb + (tid & 3) * 8;
    async_ld_b128(&((v8h*)lB[buf])[tid], src);   // LDS linear index 8*tid
  };

  stage(0, 0);
  async_wait0();
  __syncthreads();
  const int NIT = E_ / 32;
  for (int it = 0; it < NIT; ++it) {
    const int kb  = it * 32;
    const int buf = it & 1;
    if (it + 1 < NIT) stage(buf ^ 1, kb + 32);

    // A fragments: 16x32 f16, two b128 loads each from global f16 query
    const v8h* ap0 = (const v8h*)(arow0 + kb + hs * 8);
    const v8h* ap1 = (const v8h*)(arow1 + kb + hs * 8);
    const v16h a0 = cat16(ap0[0], ap0[2]);       // +16 halves = +2 v8h
    const v16h a1 = cat16(ap1[0], ap1[2]);

#pragma unroll
    for (int j = 0; j < 4; ++j) {
      const _Float16* bp = &lB[buf][(16 * j + lr) * 32 + hs * 8];
      const v16h bf = cat16(((const v8h*)bp)[0], ((const v8h*)bp)[2]);
      acc[0][j] = wmma_f16(a0, bf, acc[0][j]);
      acc[1][j] = wmma_f16(a1, bf, acc[1][j]);
    }
    async_wait0();                               // next chunk resident in LDS
    __syncthreads();
  }

  // Epilogue: bias add + scatter into Qh / Kh / Vt (f16)
#pragma unroll
  for (int j = 0; j < 4; ++j) {
    const int n = n0 + 16 * j + lr;              // output column
    const float bv = bias[n];
    const int which = n / E_;                    // 0=q 1=k 2=v (uniform per block)
    const int e = n % E_;
    const int h = e >> 6;
    const int d = e & 63;
#pragma unroll
    for (int mi = 0; mi < 2; ++mi) {
#pragma unroll
      for (int i = 0; i < 8; ++i) {
        const int m  = m0 + 16 * mi + hs * 8 + i;  // row (C layout: vgpr i = row)
        const int bb = m >> 11;                    // / S_
        const int s  = m & (S_ - 1);
        const _Float16 hv = (_Float16)(acc[mi][j][i] + bv);
        const size_t bh = (size_t)bb * H_ + h;
        if (which == 0)      Qh[(bh * S_ + s) * HD_ + d] = hv;
        else if (which == 1) Kh[(bh * S_ + s) * HD_ + d] = hv;
        else                 Vt[(bh * HD_ + d) * S_ + s] = hv;
      }
    }
  }
}

// -------------------------------------------------------------------------
// Kernel 2: flash attention. Block = 8 waves, all on the SAME (b,h) with 8
// consecutive 16-row q tiles; K and V^T chunks async-staged once per block
// into LDS (double buffered, 1 barrier / 32-key chunk). Scores computed
// transposed (S^T = K x Q^T) so each lane owns one query column.
// KEY SIMPLIFICATION: the 10*tanh clip bounds logits to [-10,10], so
// softmax needs NO running max: exp(x) spans [4.5e-5, 2.2e4] (safe in f32
// and f16), and the accumulator is never rescaled. Only a running sum is
// kept. out^T += V^T x P^T, 8 WMMAs per chunk per wave.
// -------------------------------------------------------------------------
__global__ __launch_bounds__(256) void fa_kernel(
    const _Float16* __restrict__ Qh, const _Float16* __restrict__ Kh,
    const _Float16* __restrict__ Vt, float* __restrict__ out) {
  __shared__ _Float16 lK[2][32 * 64];            // 2 x 4KB K chunk [key][d]
  __shared__ _Float16 lV[2][64 * 32];            // 2 x 4KB V^T chunk [d][key]
  const int tid  = threadIdx.x;
  const int lane = tid & 31;
  const int wave = tid >> 5;
  const int bh   = blockIdx.x >> 4;              // 16 blocks per (b,h)
  const int qt   = (blockIdx.x & 15) * 8 + wave; // 0..127
  const int h    = bh % H_;
  const int bb   = bh / H_;
  const int q0   = qt * 16;
  const int lr   = lane & 15;
  const int hs   = lane >> 4;
  const bool lo  = (lane < 16);

  const _Float16* Qp = Qh + (size_t)bh * S_ * HD_;
  const _Float16* Kp = Kh + (size_t)bh * S_ * HD_;
  const _Float16* Vp = Vt + (size_t)bh * HD_ * S_;

  // Resident Q^T B fragments (64 x 16 split into two 32-deep chunks)
  v16h bq[2];
#pragma unroll
  for (int c = 0; c < 2; ++c) {
    const v8h* qp = (const v8h*)(Qp + (size_t)(q0 + lr) * HD_ + 32 * c + hs * 8);
    bq[c] = cat16(qp[0], qp[2]);
  }

  // cooperative async K / V^T chunk stage (one b128 per thread per buffer)
  auto stage = [&](int buf, int kb) {
    async_ld_b128(&((v8h*)lK[buf])[tid], Kp + (size_t)kb * HD_ + tid * 8);
    async_ld_b128(&((v8h*)lV[buf])[tid],
                  Vp + (size_t)(tid >> 2) * S_ + kb + (tid & 3) * 8);
  };

  v8f acc[4] = {{}, {}, {}, {}};                 // out^T: 4 tiles 16(d) x 16(q)
  float lrow = 0.0f;                             // softmax denominator
  const float scale = 0.125f;                    // 1/sqrt(64)

  stage(0, 0);
  async_wait0();
  __syncthreads();
  const int NIT = S_ / 32;
  for (int it = 0; it < NIT; ++it) {
    const int buf = it & 1;
    if (it + 1 < NIT) stage(buf ^ 1, (it + 1) * 32);

    // S^T tiles: st[u][r] = S^T[key = 32*it+16u+hs*8+r][q = q0+lr]
    v8f st[2];
#pragma unroll
    for (int u = 0; u < 2; ++u) {
      v8f s = {};
#pragma unroll
      for (int c = 0; c < 2; ++c) {
        const v8h* kp =
            (const v8h*)&lK[buf][(16 * u + lr) * HD_ + 32 * c + hs * 8];
        s = wmma_f16(cat16(kp[0], kp[2]), bq[c], s);
      }
      st[u] = s;
    }

    // p = exp(10*tanh(scale*s)); bounded logits -> no max subtraction needed
    float p0[8], p1[8];
    float lsum = 0.0f;
#pragma unroll
    for (int r = 0; r < 8; ++r) {
      p0[r] = __expf(10.0f * tanhf(st[0][r] * scale));
      p1[r] = __expf(10.0f * tanhf(st[1][r] * scale));
      lsum += p0[r] + p1[r];
    }
    lrow += lsum + __shfl_xor(lsum, 16, 32);     // combine key halves L<->L+16

    // Assemble P^T B fragment (32 keys x 16 q): exchange halves, pack to f16
    float x0[8], x1[8];
#pragma unroll
    for (int r = 0; r < 8; ++r) {
      x0[r] = __shfl_xor(p0[r], 16, 32);
      x1[r] = __shfl_xor(p1[r], 16, 32);
    }
    v16h bp;
#pragma unroll
    for (int r = 0; r < 8; ++r) {
      float e0, e1;
      if (r < 4) {
        e0 = lo ? p0[2 * r]     : x1[2 * r];
        e1 = lo ? p0[2 * r + 1] : x1[2 * r + 1];
      } else {
        e0 = lo ? x0[2 * r - 8] : p1[2 * r - 8];
        e1 = lo ? x0[2 * r - 7] : p1[2 * r - 7];
      }
      bp[2 * r]     = (_Float16)e0;
      bp[2 * r + 1] = (_Float16)e1;
    }

    // out^T += V^T x P^T (V^T A fragments contiguous in LDS)
#pragma unroll
    for (int t = 0; t < 4; ++t) {
      const v8h* vp = (const v8h*)&lV[buf][(t * 16 + lr) * 32 + hs * 16];
      acc[t] = wmma_f16(cat16(vp[0], vp[1]), bp, acc[t]);
    }
    async_wait0();                               // next K/V chunk resident
    __syncthreads();
  }

  // Normalize and write: lane owns query column q0+lr, rows are d values
  const float inv = 1.0f / lrow;
  float* op = out + ((size_t)bb * S_ + q0 + lr) * E_ + h * HD_;
#pragma unroll
  for (int t = 0; t < 4; ++t)
#pragma unroll
    for (int i = 0; i < 8; ++i)
      op[t * 16 + hs * 8 + i] = acc[t][i] * inv;
}

extern "C" void kernel_launch(void* const* d_in, const int* in_sizes, int n_in,
                              void* d_out, int out_size, void* d_ws, size_t ws_size,
                              hipStream_t stream) {
  const float* query = (const float*)d_in[0];
  // d_in[1] (key) and d_in[2] (value) are ignored, per the reference
  const float* qkv_w = (const float*)d_in[3];
  const float* qkv_b = (const float*)d_in[4];
  float* out = (float*)d_out;

  const size_t nQ = (size_t)M_ * E_;             // 4M elems
  const size_t nW = (size_t)N3E * E_;            // 3M elems
  const size_t per = (size_t)B_ * H_ * S_ * HD_; // 4M halves per attn buffer

  _Float16* Qf = (_Float16*)d_ws;                // f16 query      (8 MB)
  _Float16* Wh = Qf + nQ;                        // f16 weights    (6 MB)
  _Float16* Qh = Wh + nW;                        // per-head Q     (8 MB)
  _Float16* Kh = Qh + per;                       // per-head K     (8 MB)
  _Float16* Vt = Kh + per;                       // per-head V^T   (8 MB)

  hipLaunchKernelGGL(cvt_f32_f16, dim3((int)(nQ / 4 / 256)), dim3(256), 0,
                     stream, query, Qf, (int)(nQ / 4));
  hipLaunchKernelGGL(cvt_f32_f16, dim3((int)(nW / 4 / 256)), dim3(256), 0,
                     stream, qkv_w, Wh, (int)(nW / 4));

  const int blocks1 = (M_ / 32 / 8) * (N3E / 64); // 16 * 48 = 768
  hipLaunchKernelGGL(qkv_gemm_kernel, dim3(blocks1), dim3(256), 0, stream,
                     Qf, Wh, qkv_b, Qh, Kh, Vt);

  const int blocks2 = B_ * H_ * (S_ / 16) / 8;    // 512
  hipLaunchKernelGGL(fa_kernel, dim3(blocks2), dim3(256), 0, stream,
                     Qh, Kh, Vt, out);
}
